// TransformerLayer_72430328480006
// MI455X (gfx1250) — compile-verified, matrix operand order checked by reading, not run on hardware
//
#include <hip/hip_runtime.h>
#include <hip/hip_bf16.h>
#include <cstdint>

// ---------------------------------------------------------------------------
// Fixed problem shapes (from the reference): B=2, S=2048, D=1024, H=16, DH=64,
// DM=4096.  Everything below is specialized to these.
// ---------------------------------------------------------------------------
#define BB 2
#define SS 2048
#define DD 1024
#define HH 16
#define DHH 64
#define DMM 4096

typedef __bf16 bf16;
typedef __attribute__((ext_vector_type(16))) __bf16 bf16x16;
typedef __attribute__((ext_vector_type(8)))  float  f32x8;

struct alignas(16) U128 { unsigned int u[4]; };
struct U256 { U128 lo, hi; };
struct alignas(16) F4 { float x, y, z, w; };

static_assert(sizeof(U256) == sizeof(bf16x16), "frag size mismatch");

// ---- Tensor Data Mover availability ----------------------------------------
// Device pass: probe-confirmed present (round-3 #error probe fired only for the
// host pass, device pass compiled the builtin).  Host pass falls back cleanly.
#if defined(__has_builtin)
#if __has_builtin(__builtin_amdgcn_tensor_load_to_lds) && \
    __has_builtin(__builtin_amdgcn_s_wait_tensorcnt)
#define USE_TDM 1
#endif
#endif
#ifndef USE_TDM
#define USE_TDM 0
#endif

__device__ __forceinline__ bf16 f2bf(float f) {
    unsigned int u = __builtin_bit_cast(unsigned int, f);
    unsigned int r = (u + 0x7FFFu + ((u >> 16) & 1u)) >> 16;   // RNE
    return __builtin_bit_cast(bf16, (unsigned short)r);
}

__device__ __forceinline__ f32x8 zero8() {
    f32x8 z = {0.f, 0.f, 0.f, 0.f, 0.f, 0.f, 0.f, 0.f};
    return z;
}

__device__ __forceinline__ f32x8 wmma_bf16(bf16x16 a, bf16x16 b, f32x8 c) {
    // 8-arg form: (neg_a, A, neg_b, B, c_mod, C, reuse_a, reuse_b)
    return __builtin_amdgcn_wmma_f32_16x16x32_bf16(false, a, false, b,
                                                   (short)0, c, false, false);
}

// Load one 16x32 bf16 operand fragment from a row-major LDS tile.
// A-fragment layout (ISA 7.12.2): lane L -> row (L&15); kbase = (L<16?0:8);
// v0..3 = K[kbase..kbase+7] (packed pairs, 16B contiguous), v4..7 = +16.
// B-fragments are loaded identically from a [N,K]-major ("weight layout") tile.
__device__ __forceinline__ bf16x16 ldsFrag(const bf16* base, int ld) {
    const int lane = threadIdx.x & 31;
    const bf16* p = base + (lane & 15) * ld + ((lane & 16) ? 8 : 0);
    U256 t;
    t.lo = *(const U128*)(p);
    t.hi = *(const U128*)(p + 16);
    return __builtin_bit_cast(bf16x16, t);
}

#if USE_TDM
typedef __attribute__((ext_vector_type(4))) unsigned int u32x4;
typedef __attribute__((ext_vector_type(4))) int          i32x4;
typedef __attribute__((ext_vector_type(8))) int          i32x8;
struct G0W { unsigned w[4]; };
struct G1W { unsigned w[8]; };

// Issue one TDM 2D tile load Global->LDS (per-wave op, EXEC ignored).
// D# packing per CDNA5 ISA ch.8: group0 = {count=1, lds_addr, global_addr,
// type=2}; group1 = {data_size=2B, pad ctl, tensor dims, tile dims, stride}.
// pad_interval_code/pad_amount_code reproduce the padded LDS row stride.
__device__ __forceinline__ void tdm_load_2d(const bf16* gsrc, const bf16* ldsDst,
                                            unsigned tensor_d0, unsigned tensor_d1,
                                            unsigned tile_d0, unsigned tile_d1,
                                            unsigned stride_d0,
                                            unsigned pad_interval_code,
                                            unsigned pad_amount_code) {
    const unsigned long long ga = (unsigned long long)(uintptr_t)gsrc;
    const unsigned lds = (unsigned)(uintptr_t)ldsDst;   // flat LDS addr low 32b
    G0W g0;
    g0.w[0] = 1u;                                            // count = 1 (valid)
    g0.w[1] = lds;                                           // lds_addr
    g0.w[2] = (unsigned)ga;                                  // global_addr[31:0]
    g0.w[3] = (unsigned)((ga >> 32) & 0x01FFFFFFu) | (2u << 30); // [56:32]|type=2
    G1W g1;
    g1.w[0] = (1u << 16)                       // data_size = 1 -> 2 bytes
            | (1u << 20)                       // pad_enable
            | (pad_interval_code << 22)
            | (pad_amount_code << 25);
    g1.w[1] = (tensor_d0 & 0xFFFFu) << 16;                   // tensor_dim0 lo16
    g1.w[2] = (tensor_d0 >> 16) | ((tensor_d1 & 0xFFFFu) << 16);
    g1.w[3] = (tensor_d1 >> 16) | ((tile_d0 & 0xFFFFu) << 16);
    g1.w[4] = tile_d1 & 0xFFFFu;                             // tile_dim2 = 0
    g1.w[5] = stride_d0;                                     // dim0 stride lo32
    g1.w[6] = 0u;                                            // stride hi / d1s lo
    g1.w[7] = 0u;
    const i32x4 z4 = {0, 0, 0, 0};
#if __clang_major__ >= 23
    const i32x8 z8 = {0, 0, 0, 0, 0, 0, 0, 0};
    __builtin_amdgcn_tensor_load_to_lds(__builtin_bit_cast(u32x4, g0),
                                        __builtin_bit_cast(i32x8, g1),
                                        z4, z4, z8, 0);
#else
    __builtin_amdgcn_tensor_load_to_lds(__builtin_bit_cast(u32x4, g0),
                                        __builtin_bit_cast(i32x8, g1),
                                        z4, z4, 0);
#endif
}
#endif  // USE_TDM

// ---------------------------------------------------------------------------
// fp32 -> bf16 conversion (n is always a multiple of 4 here)
// ---------------------------------------------------------------------------
__global__ __launch_bounds__(256)
void cvt_bf16_kernel(const float* __restrict__ in, bf16* __restrict__ out, int n) {
    int i = (blockIdx.x * 256 + threadIdx.x) * 4;
    if (i >= n) return;
    F4 f = *(const F4*)(in + i);
    out[i + 0] = f2bf(f.x);
    out[i + 1] = f2bf(f.y);
    out[i + 2] = f2bf(f.z);
    out[i + 3] = f2bf(f.w);
}

// ---------------------------------------------------------------------------
// Generic 128x128-tile bf16 WMMA GEMM:  C[M,N] = A[M,K] @ W[N,K]^T  (+epilogue)
// Block = 256 threads = 8 wave32s arranged 2(m) x 4(n); each wave owns 64x32.
// K-step 64: 16 v_wmma per wave between barrier pairs.
// ---------------------------------------------------------------------------
#define FLAG_RELU    1
#define FLAG_BIAS    2
#define FLAG_RESID   4
#define FLAG_ST_BF16 8
#define FLAG_ST_F32  16
#define FLAG_QKV     32
#define GLDT 72   // padded LDS row stride (elems): 144B, 16B-aligned, bank-spread

__global__ __launch_bounds__(256)
void gemm_bf16_kernel(const bf16* __restrict__ A, const bf16* __restrict__ W,
                      int M, int N, int K, int flags,
                      const float* __restrict__ bias,
                      const float* __restrict__ resid,
                      bf16* __restrict__ outb, float* __restrict__ outf) {
    __shared__ __align__(16) bf16 As[128 * GLDT];   // 18 KB
    __shared__ __align__(16) bf16 Ws[128 * GLDT];   // 18 KB

    const int tid  = threadIdx.x;
    const int wave = tid >> 5;
    const int lane = tid & 31;
    const int m0 = blockIdx.y * 128;
    const int n0 = blockIdx.x * 128;
    const int wm = (wave & 1) * 64;      // wave m offset within tile
    const int wn = (wave >> 1) * 32;     // wave n offset within tile
    const int lr = tid >> 3;             // loader row 0..31
    const int lc = (tid & 7) * 8;        // loader col {0,8,...,56}

    f32x8 acc[4][2];
#pragma unroll
    for (int mt = 0; mt < 4; ++mt) { acc[mt][0] = zero8(); acc[mt][1] = zero8(); }

    for (int k0 = 0; k0 < K; k0 += 64) {
        // Stage 128x64 A and W tiles: issue all 8 global b128 loads first so
        // they are simultaneously in flight, then the 8 ds_store_b128.
        U128 ta[4], tw[4];
#pragma unroll
        for (int p = 0; p < 4; ++p) {
            const int r = lr + p * 32;
            ta[p] = *(const U128*)&A[(size_t)(m0 + r) * K + k0 + lc];
            tw[p] = *(const U128*)&W[(size_t)(n0 + r) * K + k0 + lc];
        }
#pragma unroll
        for (int p = 0; p < 4; ++p) {
            const int r = lr + p * 32;
            *(U128*)&As[r * GLDT + lc] = ta[p];
            *(U128*)&Ws[r * GLDT + lc] = tw[p];
        }
        if (k0 + 64 < K) {   // software prefetch of next K tile -> global_prefetch_b8
            __builtin_prefetch(&A[(size_t)(m0 + lr) * K + k0 + 64 + lc], 0, 1);
            __builtin_prefetch(&W[(size_t)(n0 + lr) * K + k0 + 64 + lc], 0, 1);
        }
        __syncthreads();

        bf16x16 b[2][2];
#pragma unroll
        for (int nt = 0; nt < 2; ++nt) {
            b[nt][0] = ldsFrag(&Ws[(wn + nt * 16) * GLDT + 0],  GLDT);
            b[nt][1] = ldsFrag(&Ws[(wn + nt * 16) * GLDT + 32], GLDT);
        }
#pragma unroll
        for (int mt = 0; mt < 4; ++mt) {
            bf16x16 a0 = ldsFrag(&As[(wm + mt * 16) * GLDT + 0],  GLDT);
            bf16x16 a1 = ldsFrag(&As[(wm + mt * 16) * GLDT + 32], GLDT);
            acc[mt][0] = wmma_bf16(a0, b[0][0], acc[mt][0]);
            acc[mt][0] = wmma_bf16(a1, b[0][1], acc[mt][0]);
            acc[mt][1] = wmma_bf16(a0, b[1][0], acc[mt][1]);
            acc[mt][1] = wmma_bf16(a1, b[1][1], acc[mt][1]);
        }
        __syncthreads();
    }

    // Epilogue.  C layout: lane -> col (lane&15), vgpr i -> row i + (lane<16?0:8).
    const int colL = lane & 15;
    const int rowH = (lane & 16) ? 8 : 0;
#pragma unroll
    for (int mt = 0; mt < 4; ++mt) {
#pragma unroll
        for (int nt = 0; nt < 2; ++nt) {
            const int gn = n0 + wn + nt * 16 + colL;
#pragma unroll
            for (int i = 0; i < 8; ++i) {
                const int gm = m0 + wm + mt * 16 + rowH + i;
                float v = acc[mt][nt][i];
                if (flags & FLAG_BIAS) v += bias[gn];
                if (flags & FLAG_RELU) v = fmaxf(v, 0.0f);
                if (flags & FLAG_QKV) {
                    // m = b*S + s ; n = h*DH + e ; scatter to [B,H,S,DH]
                    const int b2 = gm >> 11, s = gm & (SS - 1);
                    const int h = gn >> 6,  e = gn & (DHH - 1);
                    outb[(((size_t)(b2 * HH + h) * SS) + s) * DHH + e] = f2bf(v);
                } else {
                    const size_t idx = (size_t)gm * N + gn;
                    if (flags & FLAG_RESID)   v += resid[idx];
                    if (flags & FLAG_ST_F32)  outf[idx] = v;
                    if (flags & FLAG_ST_BF16) outb[idx] = f2bf(v);
                }
            }
        }
    }
}

// ---------------------------------------------------------------------------
// Flash attention (causal), one workgroup (4 waves) per (b*H+h, 64-row Qblock).
// Each wave owns a 16-row strip.  All matmuls via WMMA.  Q/K tiles are staged
// with the Tensor Data Mover (pad_enable reproduces the 72-elem LDS stride:
// pad_interval = 32 DWORDs (code 4), pad_amount = 4 DWORDs (code 3)).
// ---------------------------------------------------------------------------
#define ALD 72   // padded LDS row stride: 144B, 16B aligned, bank-spread

__global__ __launch_bounds__(128)
void flash_attn_kernel(const bf16* __restrict__ Q, const bf16* __restrict__ Kb,
                       const bf16* __restrict__ Vb, bf16* __restrict__ O) {
    __shared__ __align__(16) bf16 Qs [64 * ALD];  // [row][dh]
    __shared__ __align__(16) bf16 Ks [64 * ALD];  // [key][dh]   (weight-layout for QK^T)
    __shared__ __align__(16) bf16 VTs[64 * ALD];  // [dh][key]   (weight-layout for P@V)
    __shared__ __align__(16) bf16 Ps [64 * ALD];  // [row][key]  per-wave staging for P

    const int tid  = threadIdx.x;
    const int wave = tid >> 5;
    const int lane = tid & 31;
    const int qb   = blockIdx.x;                 // 0..S/64-1
    const int bh   = blockIdx.y;                 // b*H + h
    const size_t base = (size_t)bh * SS * DHH;

    // load Q tile (64x64)
#if USE_TDM
    if (wave == 0)
        tdm_load_2d(Q + base + (size_t)(qb * 64) * DHH, Qs,
                    /*tensor*/64, 64, /*tile*/64, 64, /*stride*/64,
                    /*pad_interval 32DW*/4, /*pad_amount 4DW*/3);
#else
    {
        const int c = (tid & 7) * 8;
        for (int r = tid >> 3; r < 64; r += 16)
            *(U128*)&Qs[r * ALD + c] =
                *(const U128*)&Q[base + (size_t)(qb * 64 + r) * DHH + c];
    }
#endif

    const int wrow = wave * 16;
    const int colL = lane & 15;
    const int rowH = (lane & 16) ? 8 : 0;
    const int grow0 = qb * 64 + wrow + rowH;     // global row of vgpr i=0

    f32x8 o[4];
    float mrow[8], lrow[8];
#pragma unroll
    for (int d = 0; d < 4; ++d) o[d] = zero8();
#pragma unroll
    for (int i = 0; i < 8; ++i) { mrow[i] = -1e30f; lrow[i] = 0.f; }

    for (int jt = 0; jt <= qb; ++jt) {           // uniform trip count per block
        __syncthreads();                         // K/V tiles free to overwrite
#if USE_TDM
        if (wave == 0)                           // K tile via TDM: [key][dh]
            tdm_load_2d(Kb + base + (size_t)(jt * 64) * DHH, Ks,
                        64, 64, 64, 64, 64, 4, 3);
#else
        {   // K tile: direct copy [key][dh]
            const int c = (tid & 7) * 8;
            for (int r = tid >> 3; r < 64; r += 16)
                *(U128*)&Ks[r * ALD + c] =
                    *(const U128*)&Kb[base + (size_t)(jt * 64 + r) * DHH + c];
        }
#endif
        {   // V tile: transpose into [dh][key] (TDM cannot transpose)
            union { U128 u; bf16 h[8]; } vv;
            const int c = (tid & 7) * 8;
            for (int r = tid >> 3; r < 64; r += 16) {
                vv.u = *(const U128*)&Vb[base + (size_t)(jt * 64 + r) * DHH + c];
#pragma unroll
                for (int i = 0; i < 8; ++i) VTs[(c + i) * ALD + r] = vv.h[i];
            }
        }
#if USE_TDM
        if (wave == 0) __builtin_amdgcn_s_wait_tensorcnt(0);  // Q (+this K) done
#endif
        __syncthreads();

        // ---- S = Q @ K^T (16x64 per wave): 8 WMMAs ----
        bf16x16 aq0 = ldsFrag(&Qs[wrow * ALD + 0],  ALD);
        bf16x16 aq1 = ldsFrag(&Qs[wrow * ALD + 32], ALD);
        f32x8 s[4];
#pragma unroll
        for (int nt = 0; nt < 4; ++nt) {
            f32x8 c = zero8();
            c = wmma_bf16(aq0, ldsFrag(&Ks[nt * 16 * ALD + 0],  ALD), c);
            c = wmma_bf16(aq1, ldsFrag(&Ks[nt * 16 * ALD + 32], ALD), c);
            s[nt] = c;
        }

        // ---- scale + causal mask ----
        const float scale = 0.125f;              // 1/sqrt(64)
#pragma unroll
        for (int nt = 0; nt < 4; ++nt) {
            const int gcol = jt * 64 + nt * 16 + colL;
#pragma unroll
            for (int i = 0; i < 8; ++i) {
                float v = s[nt][i] * scale;
                if (gcol > grow0 + i) v = -1e30f;
                s[nt][i] = v;
            }
        }

        // ---- online softmax (row stats across 16-lane halves) ----
#pragma unroll
        for (int i = 0; i < 8; ++i) {
            float rm = fmaxf(fmaxf(s[0][i], s[1][i]), fmaxf(s[2][i], s[3][i]));
#pragma unroll
            for (int mk = 1; mk <= 8; mk <<= 1) rm = fmaxf(rm, __shfl_xor(rm, mk, 32));
            const float nm = fmaxf(mrow[i], rm);
            const float alpha = __expf(mrow[i] - nm);
            float ps = 0.f;
#pragma unroll
            for (int nt = 0; nt < 4; ++nt) {
                float p = __expf(s[nt][i] - nm);
                s[nt][i] = p;
                ps += p;
            }
#pragma unroll
            for (int mk = 1; mk <= 8; mk <<= 1) ps += __shfl_xor(ps, mk, 32);
            lrow[i] = lrow[i] * alpha + ps;
            mrow[i] = nm;
#pragma unroll
            for (int d = 0; d < 4; ++d) o[d][i] *= alpha;
        }

        // ---- stage P (bf16) through this wave's private LDS strip ----
#pragma unroll
        for (int nt = 0; nt < 4; ++nt)
#pragma unroll
            for (int i = 0; i < 8; ++i)
                Ps[(wrow + rowH + i) * ALD + nt * 16 + colL] = f2bf(s[nt][i]);
        // same-wave LDS ops are in-order; compiler inserts the dscnt wait.

        // ---- O += P @ V : 8 WMMAs ----
        bf16x16 ap0 = ldsFrag(&Ps[wrow * ALD + 0],  ALD);
        bf16x16 ap1 = ldsFrag(&Ps[wrow * ALD + 32], ALD);
#pragma unroll
        for (int dt = 0; dt < 4; ++dt) {
            o[dt] = wmma_bf16(ap0, ldsFrag(&VTs[dt * 16 * ALD + 0],  ALD), o[dt]);
            o[dt] = wmma_bf16(ap1, ldsFrag(&VTs[dt * 16 * ALD + 32], ALD), o[dt]);
        }
    }

    // ---- normalize and store O (bf16, [B,H,S,DH]) ----
#pragma unroll
    for (int i = 0; i < 8; ++i) {
        const float inv = 1.0f / lrow[i];
        const size_t rbase = base + (size_t)(qb * 64 + wrow + rowH + i) * DHH;
#pragma unroll
        for (int dt = 0; dt < 4; ++dt)
            O[rbase + dt * 16 + colL] = f2bf(o[dt][i] * inv);
    }
}

// ---------------------------------------------------------------------------
// Per-head output projection + attention residual:
// x1[b,s,h*64+f] = x[b,s,h*64+f] + sum_e O[b,h,s,e]*Wo[h,f,e]
// One workgroup (4 waves) per (64-row s-block, b*H+h).  Writes fp32 + bf16.
// ---------------------------------------------------------------------------
__global__ __launch_bounds__(128)
void oproj_kernel(const bf16* __restrict__ O, const bf16* __restrict__ Wo,
                  const float* __restrict__ x,
                  float* __restrict__ x1f, bf16* __restrict__ x1b) {
    __shared__ __align__(16) bf16 Os[64 * ALD];   // [row(s)][e]
    __shared__ __align__(16) bf16 Ws[64 * ALD];   // [f][e]  (weight layout)

    const int tid  = threadIdx.x;
    const int wave = tid >> 5;
    const int lane = tid & 31;
    const int sb   = blockIdx.x;                 // 0..S/64-1
    const int bh   = blockIdx.y;
    const int b = bh >> 4, h = bh & (HH - 1);

    {
        const int c = (tid & 7) * 8;
        for (int r = tid >> 3; r < 64; r += 16) {
            *(U128*)&Os[r * ALD + c] =
                *(const U128*)&O[((size_t)bh * SS + sb * 64 + r) * DHH + c];
            *(U128*)&Ws[r * ALD + c] =
                *(const U128*)&Wo[(size_t)h * DHH * DHH + (size_t)r * DHH + c];
        }
    }
    __syncthreads();

    const int wrow = wave * 16;
    const int colL = lane & 15;
    const int rowH = (lane & 16) ? 8 : 0;

    bf16x16 a0 = ldsFrag(&Os[wrow * ALD + 0],  ALD);
    bf16x16 a1 = ldsFrag(&Os[wrow * ALD + 32], ALD);
#pragma unroll
    for (int nt = 0; nt < 4; ++nt) {
        f32x8 c = zero8();
        c = wmma_bf16(a0, ldsFrag(&Ws[nt * 16 * ALD + 0],  ALD), c);
        c = wmma_bf16(a1, ldsFrag(&Ws[nt * 16 * ALD + 32], ALD), c);
#pragma unroll
        for (int i = 0; i < 8; ++i) {
            const size_t idx = ((size_t)b * SS + sb * 64 + wrow + rowH + i) * DD
                             + h * DHH + nt * 16 + colL;
            const float v = c[i] + x[idx];
            x1f[idx] = v;
            x1b[idx] = f2bf(v);
        }
    }
}

// ---------------------------------------------------------------------------
// Host launcher
// ---------------------------------------------------------------------------
extern "C" void kernel_launch(void* const* d_in, const int* in_sizes, int n_in,
                              void* d_out, int out_size, void* d_ws, size_t ws_size,
                              hipStream_t stream) {
    (void)in_sizes; (void)n_in; (void)out_size; (void)ws_size;
    const float* x     = (const float*)d_in[0];
    const float* Wq    = (const float*)d_in[1];
    const float* Wk    = (const float*)d_in[2];
    const float* Wv    = (const float*)d_in[3];
    const float* Wo    = (const float*)d_in[4];
    const float* W_in  = (const float*)d_in[5];
    const float* b_in  = (const float*)d_in[6];
    const float* W_out = (const float*)d_in[7];
    const float* b_out = (const float*)d_in[8];
    float* out = (float*)d_out;

    char* ws = (char*)d_ws;
    const size_t MB = 1u << 20;
    // Workspace layout (~64 MiB total, with reuse):
    bf16* Qb    = (bf16*)(ws + 0 * MB);    // 8 MiB  } reused as hmid (32 MiB)
    bf16* Kbuf  = (bf16*)(ws + 8 * MB);    // 8 MiB  } after attention+oproj
    bf16* Vbuf  = (bf16*)(ws + 16 * MB);   // 8 MiB  }
    bf16* Ob    = (bf16*)(ws + 24 * MB);   // 8 MiB  }
    bf16* hmid  = (bf16*)(ws + 0 * MB);    // 32 MiB (reuse of Qb..Ob)
    bf16* xb    = (bf16*)(ws + 32 * MB);   // 8 MiB, dead after QKV
    bf16* x1b   = (bf16*)(ws + 32 * MB);   // reuse of xb
    bf16* Wqb   = (bf16*)(ws + 40 * MB);   // 2 MiB
    bf16* Wkb   = (bf16*)(ws + 42 * MB);   // 2 MiB
    bf16* Wvb   = (bf16*)(ws + 44 * MB);   // 2 MiB
    bf16* Wob   = (bf16*)(ws + 46 * MB);   // 128 KiB
    bf16* Winb  = (bf16*)(ws + 48 * MB);   // 8 MiB
    bf16* Woutb = (bf16*)(ws + 56 * MB);   // 8 MiB

    auto cvt = [&](const float* src, bf16* dst, int n) {
        cvt_bf16_kernel<<<dim3((n / 4 + 255) / 256), dim3(256), 0, stream>>>(src, dst, n);
    };
    cvt(x,     xb,    BB * SS * DD);
    cvt(Wq,    Wqb,   HH * DHH * DD);
    cvt(Wk,    Wkb,   HH * DHH * DD);
    cvt(Wv,    Wvb,   HH * DHH * DD);
    cvt(Wo,    Wob,   HH * DHH * DHH);
    cvt(W_in,  Winb,  DMM * DD);
    cvt(W_out, Woutb, DD * DMM);

    const int M = BB * SS;   // 4096
    // Q/K/V projections: [4096,1024] @ [1024,1024]^T -> scattered [B,H,S,DH]
    dim3 gQKV(DD / 128, M / 128);
    gemm_bf16_kernel<<<gQKV, 256, 0, stream>>>(xb, Wqb, M, DD, DD, FLAG_QKV,
                                               nullptr, nullptr, Qb, nullptr);
    gemm_bf16_kernel<<<gQKV, 256, 0, stream>>>(xb, Wkb, M, DD, DD, FLAG_QKV,
                                               nullptr, nullptr, Kbuf, nullptr);
    gemm_bf16_kernel<<<gQKV, 256, 0, stream>>>(xb, Wvb, M, DD, DD, FLAG_QKV,
                                               nullptr, nullptr, Vbuf, nullptr);

    // Fused causal flash attention
    flash_attn_kernel<<<dim3(SS / 64, BB * HH), 128, 0, stream>>>(Qb, Kbuf, Vbuf, Ob);

    // Per-head O-projection + residual; fp32 residual stream lands in d_out
    oproj_kernel<<<dim3(SS / 64, BB * HH), 128, 0, stream>>>(Ob, Wob, x, out, x1b);

    // MLP up: relu(x1 @ W_in^T + b_in) -> bf16 hmid
    gemm_bf16_kernel<<<dim3(DMM / 128, M / 128), 256, 0, stream>>>(
        x1b, Winb, M, DMM, DD, FLAG_RELU | FLAG_BIAS | FLAG_ST_BF16,
        b_in, nullptr, hmid, nullptr);

    // MLP down + residual + bias -> fp32 d_out (in-place residual read is
    // per-element read-before-write within one lane, safe)
    gemm_bf16_kernel<<<dim3(DD / 128, M / 128), 256, 0, stream>>>(
        hmid, Woutb, M, DD, DMM, FLAG_BIAS | FLAG_RESID | FLAG_ST_F32,
        b_out, out, nullptr, out);
}
